// FusedMoEModule_12094627905713
// MI455X (gfx1250) — matrix operand order repaired and checked by reading, not compiled, not run on hardware
//
#include <hip/hip_runtime.h>
#include <hip/hip_bf16.h>
#include <math.h>

#define N_TOK 1024
#define TOPK  2
#define NEXP  16
#define HDIM  1024
#define DDIM  2048
#define RDIM  16
#define NK    (N_TOK * TOPK)
#define LORA_SCALE 0.25f

typedef __attribute__((ext_vector_type(16))) __bf16 v16bf;
typedef __attribute__((ext_vector_type(8)))  float  v8f;

union Frag { v16bf v; uint4 q[2]; };

__device__ __forceinline__ unsigned short f2bf(float f) {
  union { float f; unsigned u; } v; v.f = f;
  unsigned r = v.u + 0x7FFFu + ((v.u >> 16) & 1u);   // round-to-nearest-even
  return (unsigned short)(r >> 16);
}
__device__ __forceinline__ float bf2f(unsigned short h) {
  union { unsigned u; float f; } v; v.u = ((unsigned)h) << 16;
  return v.f;
}

#define WMMA_BF16(A_, B_, C_) \
  __builtin_amdgcn_wmma_f32_16x16x32_bf16(false, (A_), false, (B_), (short)0, (C_), false, false)

// pack two floats to adjacent bf16 (lo, hi)
#define PACK2(lo_, hi_) (((unsigned)f2bf(hi_) << 16) | (unsigned)f2bf(lo_))

// ---------------- small kernels ----------------

__global__ void zero_counts_kernel(int* counts) {
  if (threadIdx.x < NEXP) counts[threadIdx.x] = 0;
}

__global__ void route_kernel(const int* __restrict__ topk_ids,
                             int* __restrict__ counts,
                             int* __restrict__ rowmap) {
  int i = blockIdx.x * blockDim.x + threadIdx.x;
  if (i < NK) {
    int e = topk_ids[i];
    int p = atomicAdd(&counts[e], 1);
    rowmap[e * NK + p] = i;   // slot order within bucket does not affect results
  }
}

__global__ void cvt_bf16_kernel(const float* __restrict__ x,
                                unsigned short* __restrict__ xb, int n) {
  int i = blockIdx.x * blockDim.x + threadIdx.x;
  if (i < n) xb[i] = f2bf(x[i]);
}

// t[nk,r] = sum_d x[n,d] * up_a[e,r,d]; block = one routed row, 16 lanes per r.
__global__ __launch_bounds__(256) void lora_t_up_kernel(
    const float* __restrict__ x, const float* __restrict__ up_a,
    const int* __restrict__ topk_ids, float* __restrict__ t) {
  int nk = blockIdx.x;
  int e  = topk_ids[nk];
  int r  = threadIdx.x >> 4;
  int l  = threadIdx.x & 15;
  const float* xr = x + (size_t)(nk >> 1) * DDIM;          // TOPK == 2
  const float* ar = up_a + ((size_t)e * RDIM + r) * DDIM;
  float s = 0.f;
  for (int d = l; d < DDIM; d += 16) s += xr[d] * ar[d];   // 64B coalesced per 16 lanes
  for (int m = 8; m >= 1; m >>= 1) s += __shfl_xor(s, m, 32);
  if (l == 0) t[nk * RDIM + r] = s;
}

// t2[nk,r] = sum_h act[nk,h] * down_a[e,r,h]
__global__ __launch_bounds__(256) void lora_t_down_kernel(
    const unsigned short* __restrict__ act, const float* __restrict__ down_a,
    const int* __restrict__ topk_ids, float* __restrict__ t2) {
  int nk = blockIdx.x;
  int e  = topk_ids[nk];
  int r  = threadIdx.x >> 4;
  int l  = threadIdx.x & 15;
  const unsigned short* ar = act + (size_t)nk * HDIM;
  const float* dr = down_a + ((size_t)e * RDIM + r) * HDIM;
  float s = 0.f;
  for (int h = l; h < HDIM; h += 16) s += bf2f(ar[h]) * dr[h];
  for (int m = 8; m >= 1; m >>= 1) s += __shfl_xor(s, m, 32);
  if (l == 0) t2[nk * RDIM + r] = s;
}

// act = gelu_erf(up[:, :H]) * up[:, H:]
__global__ void act_kernel(const unsigned short* __restrict__ up,
                           unsigned short* __restrict__ act) {
  int i = blockIdx.x * blockDim.x + threadIdx.x;
  if (i >= NK * HDIM) return;
  int nk = i / HDIM, h = i - nk * HDIM;
  float a = bf2f(up[(size_t)nk * (2 * HDIM) + h]);
  float b = bf2f(up[(size_t)nk * (2 * HDIM) + HDIM + h]);
  float g = 0.5f * a * (1.f + erff(a * 0.70710678118654752f));
  act[i] = f2bf(g * b);
}

// out[n,d] = down[(n,0),d] + down[(n,1),d]   (deterministic moe_sum, no atomics)
__global__ void moe_sum_kernel(const float* __restrict__ down,
                               float* __restrict__ out) {
  int i = blockIdx.x * blockDim.x + threadIdx.x;
  if (i >= N_TOK * DDIM) return;
  int n = i / DDIM, d = i - n * DDIM;
  out[i] = down[(size_t)(n * 2) * DDIM + d] + down[(size_t)(n * 2 + 1) * DDIM + d];
}

// ---------------- grouped WMMA GEMM ----------------
// C[row, col] = sum_k A_bf16[row, k] * B_f32[e, col, k]  (+ LoRA epilogue)
// Block tile: 64 rows x 128 cols, K chunk 64; 8 waves (2m x 4n), wave tile 32x32
// (4 accumulators, 4 WMMA per 4 ds_load_b128 for A + 32 16-bit LDS reads for B). A tile arrives via async DMA
// (ASYNCcnt); B tile is converted f32->bf16 and transposed into LDS with
// packed ds_store_b64 (4-row x 4-k micro-tiles per thread).

template<int KDIM, int NOUT, bool DOWN>
__global__ __launch_bounds__(256) void moe_gemm_kernel(
    const unsigned short* __restrict__ A,     // bf16 [*, KDIM]
    const float* __restrict__ Bsrc,           // [E, NOUT, KDIM]
    const int* __restrict__ counts,
    const int* __restrict__ rowmap,
    const float* __restrict__ lt,             // [NK, R]
    const float* __restrict__ loraB,          // [E, NOUT, R]
    const float* __restrict__ tw,             // [NK] (down only)
    unsigned short* __restrict__ outH,        // up path (bf16)
    float* __restrict__ outF)                 // down path (f32)
{
  constexpr int MT = 64, NT = 128, KT = 64;
  const int e     = blockIdx.y / (NK / MT);
  const int mtile = blockIdx.y % (NK / MT);
  const int cnt   = counts[e];
  if (mtile * MT >= cnt) return;
  const int n0 = blockIdx.x * NT;

  __shared__ __align__(16) unsigned short As[MT][KT];        // 8 KB
  __shared__ __align__(16) unsigned short Bs[KT][NT + 8];    // ~17 KB, transposed [k][n]
  __shared__ int   rows[MT];
  __shared__ float tL[MT][RDIM];

  const int tid = threadIdx.x;
  if (tid < MT) {
    int gr = mtile * MT + tid;
    int cl = gr < cnt ? gr : cnt - 1;
    rows[tid] = (gr < cnt) ? rowmap[e * NK + cl] : -1;
  }
  for (int i = tid; i < MT * RDIM; i += 256) {
    int r = i >> 4, c = i & 15;
    int gr = mtile * MT + r;
    int cl = gr < cnt ? gr : cnt - 1;
    int nk = rowmap[e * NK + cl];
    tL[r][c] = lt[(size_t)nk * RDIM + c];
  }
  __syncthreads();

  const float* Bp = Bsrc + ((size_t)e * NOUT + n0) * KDIM;
  v8f acc00 = {}, acc01 = {}, acc10 = {}, acc11 = {};
  const int lane = tid & 31;
  const int wave = tid >> 5;
  const int mw = wave & 1;        // 0..1  (32-row half)
  const int nw = wave >> 1;       // 0..3  (32-col group)
  const int k4  = (tid & 15) * 4; // B loader: 4-k group (covers 64 k across 16 lanes)
  const int nr0 = (tid >> 4) * 4; // B loader: 4-row group base (covers 64 rows; x2 iters)

  for (int k0 = 0; k0 < KDIM; k0 += KT) {
    { // A tile 64x64 bf16: async DMA to LDS (ASYNCcnt path), 2x b128 per thread
      int r = tid >> 2, cs = (tid & 3) * 16;
      int nk = rows[r];
      size_t arow = (nk < 0) ? 0 : (DOWN ? (size_t)nk : (size_t)(nk >> 1));
      const unsigned short* gsrc = A + arow * KDIM + k0 + cs;
      unsigned lds0 = (unsigned)(uintptr_t)&As[r][cs];
      unsigned long long ga0 = (unsigned long long)gsrc;
      asm volatile("global_load_async_to_lds_b128 %0, %1, off"
                   :: "v"(lds0), "v"(ga0) : "memory");
      unsigned lds1 = lds0 + 16u;
      unsigned long long ga1 = ga0 + 16ull;
      asm volatile("global_load_async_to_lds_b128 %0, %1, off"
                   :: "v"(lds1), "v"(ga1) : "memory");
    }
    // B tile 128n x 64k f32 -> bf16 transposed into Bs[k][n].
    // Each thread: 2 iterations of a 4-row x 4-k micro-tile; 16 lanes sweep a
    // 256B-contiguous k-row per global load; stores are packed ds_store_b64.
    #pragma unroll
    for (int it = 0; it < 2; ++it) {
      int nq = it * 64 + nr0;
      const float* s0 = Bp + (size_t)nq * KDIM + k0 + k4;
      float4 f0 = *(const float4*)(s0);
      float4 f1 = *(const float4*)(s0 + KDIM);
      float4 f2 = *(const float4*)(s0 + 2 * KDIM);
      float4 f3 = *(const float4*)(s0 + 3 * KDIM);
      uint2 q0 = { PACK2(f0.x, f1.x), PACK2(f2.x, f3.x) };
      uint2 q1 = { PACK2(f0.y, f1.y), PACK2(f2.y, f3.y) };
      uint2 q2 = { PACK2(f0.z, f1.z), PACK2(f2.z, f3.z) };
      uint2 q3 = { PACK2(f0.w, f1.w), PACK2(f2.w, f3.w) };
      *(uint2*)&Bs[k4 + 0][nq] = q0;
      *(uint2*)&Bs[k4 + 1][nq] = q1;
      *(uint2*)&Bs[k4 + 2][nq] = q2;
      *(uint2*)&Bs[k4 + 3][nq] = q3;
    }
    if (k0 + KT < KDIM)
      __builtin_prefetch(Bp + (size_t)nr0 * KDIM + k0 + KT + k4, 0, 1);
    asm volatile("s_wait_asynccnt 0x0" ::: "memory");
    __syncthreads();

    #pragma unroll
    for (int kk = 0; kk < KT; kk += 32) {
      Frag a0, a1, b0, b1;
      // A frag: lane L -> row (L&15); K halves split by L>=16
      int arl = lane & 15;
      int ak  = kk + (lane >> 4) * 8;
      a0.q[0] = *(const uint4*)&As[mw * 32 + arl][ak];
      a0.q[1] = *(const uint4*)&As[mw * 32 + arl][ak + 16];
      a1.q[0] = *(const uint4*)&As[mw * 32 + 16 + arl][ak];
      a1.q[1] = *(const uint4*)&As[mw * 32 + 16 + arl][ak + 16];
      // B frag: lane L -> n = (L&15); k in table order matching A (kk + 8*(L>=16) + s, +16 for s>=8)
      int bk = kk + (lane >> 4) * 8;
      int bn = nw * 32 + (lane & 15);
      unsigned w0[8], w1[8];
      #pragma unroll
      for (int p = 0; p < 8; ++p) {
        int ka = bk + 2 * p + (p >= 4 ? 8 : 0);
        w0[p] = (unsigned)Bs[ka][bn] | ((unsigned)Bs[ka + 1][bn] << 16);
        w1[p] = (unsigned)Bs[ka][bn + 16] | ((unsigned)Bs[ka + 1][bn + 16] << 16);
      }
      b0.q[0] = make_uint4(w0[0], w0[1], w0[2], w0[3]);
      b0.q[1] = make_uint4(w0[4], w0[5], w0[6], w0[7]);
      b1.q[0] = make_uint4(w1[0], w1[1], w1[2], w1[3]);
      b1.q[1] = make_uint4(w1[4], w1[5], w1[6], w1[7]);
      acc00 = WMMA_BF16(a0.v, b0.v, acc00);
      acc01 = WMMA_BF16(a0.v, b1.v, acc01);
      acc10 = WMMA_BF16(a1.v, b0.v, acc10);
      acc11 = WMMA_BF16(a1.v, b1.v, acc11);
    }
    __syncthreads();
  }

  // Epilogue: C layout -> element (m = v + 8*(lane>=16), n = lane&15)
  const int lrow_hi = (lane >> 4) * 8;
  const int ncb = n0 + nw * 32;
  v8f accs[2][2] = { { acc00, acc01 }, { acc10, acc11 } };
  #pragma unroll
  for (int tm = 0; tm < 2; ++tm) {
    #pragma unroll
    for (int tn = 0; tn < 2; ++tn) {
      #pragma unroll
      for (int v = 0; v < 8; ++v) {
        int ml  = mw * 32 + tm * 16 + lrow_hi + v;
        int nk  = rows[ml];
        int col = ncb + tn * 16 + (lane & 15);
        float val = accs[tm][tn][v];
        const float* br = loraB + ((size_t)e * NOUT + col) * RDIM;
        float l = 0.f;
        #pragma unroll
        for (int r = 0; r < RDIM; ++r) l += tL[ml][r] * br[r];
        val += LORA_SCALE * l;
        if (nk >= 0) {
          if (DOWN) outF[(size_t)nk * NOUT + col] = val * tw[nk];
          else      outH[(size_t)nk * NOUT + col] = f2bf(val);
        }
      }
    }
  }
}

// ---------------- launch ----------------

extern "C" void kernel_launch(void* const* d_in, const int* in_sizes, int n_in,
                              void* d_out, int out_size, void* d_ws, size_t ws_size,
                              hipStream_t stream) {
  const float* hidden  = (const float*)d_in[0];
  const float* tw      = (const float*)d_in[1];
  const int*   ids     = (const int*)  d_in[2];
  const float* w_up    = (const float*)d_in[3];
  const float* w_down  = (const float*)d_in[4];
  const float* up_a    = (const float*)d_in[5];
  const float* up_b    = (const float*)d_in[6];
  const float* down_a  = (const float*)d_in[7];
  const float* down_b  = (const float*)d_in[8];
  float* out = (float*)d_out;

  char* ws = (char*)d_ws;
  size_t off = 0;
  int* counts = (int*)(ws + off);                  off += 256;
  int* rowmap = (int*)(ws + off);                  off += (size_t)NEXP * NK * 4;
  unsigned short* xb   = (unsigned short*)(ws + off); off += (size_t)N_TOK * DDIM * 2;
  float* lt            = (float*)(ws + off);          off += (size_t)NK * RDIM * 4;
  unsigned short* upb  = (unsigned short*)(ws + off); off += (size_t)NK * 2 * HDIM * 2;
  unsigned short* actb = (unsigned short*)(ws + off); off += (size_t)NK * HDIM * 2;
  float* lt2           = (float*)(ws + off);          off += (size_t)NK * RDIM * 4;
  float* downb         = (float*)(ws + off);          off += (size_t)NK * DDIM * 4;
  (void)off; (void)ws_size; (void)in_sizes; (void)n_in; (void)out_size;

  // 1) routing
  zero_counts_kernel<<<1, 64, 0, stream>>>(counts);
  route_kernel<<<(NK + 255) / 256, 256, 0, stream>>>(ids, counts, rowmap);
  // 2) x -> bf16
  cvt_bf16_kernel<<<(N_TOK * DDIM + 255) / 256, 256, 0, stream>>>(hidden, xb, N_TOK * DDIM);
  // 3) LoRA intermediate t = x @ up_a^T  (one block per routed row)
  lora_t_up_kernel<<<NK, 256, 0, stream>>>(hidden, up_a, ids, lt);
  // 4) up GEMM (+LoRA) -> up buffer [NK, 2H] bf16
  {
    dim3 grid(2 * HDIM / 128, NEXP * (NK / 64));
    moe_gemm_kernel<DDIM, 2 * HDIM, false><<<grid, 256, 0, stream>>>(
        xb, w_up, counts, rowmap, lt, up_b, nullptr, upb, nullptr);
  }
  // 5) gelu(first half) * second half -> act [NK, H] bf16
  act_kernel<<<(NK * HDIM + 255) / 256, 256, 0, stream>>>(upb, actb);
  // 6) LoRA intermediate t2 = act @ down_a^T
  lora_t_down_kernel<<<NK, 256, 0, stream>>>(actb, down_a, ids, lt2);
  // 7) down GEMM (+LoRA, * routed weight) -> down buffer [NK, D] f32
  {
    dim3 grid(DDIM / 128, NEXP * (NK / 64));
    moe_gemm_kernel<HDIM, DDIM, true><<<grid, 256, 0, stream>>>(
        actb, w_down, counts, rowmap, lt2, down_b, tw, nullptr, downb);
  }
  // 8) deterministic top-k sum -> out [N, D]
  moe_sum_kernel<<<(N_TOK * DDIM + 255) / 256, 256, 0, stream>>>(downb, out);
}
